// KarmaBlock_7335804142293
// MI455X (gfx1250) — matrix operand order repaired and checked by reading, compiled, not run on hardware
//
#include <hip/hip_runtime.h>
#include <hip/hip_bf16.h>
#include <cstdint>
#include <cstddef>

// ---------------------------------------------------------------------------
// Types for CDNA5 WMMA (wave32, 16x16x32 bf16, f32 accum) + TDM descriptors
// ---------------------------------------------------------------------------
typedef __attribute__((ext_vector_type(16))) __bf16 v16bf;
typedef __attribute__((ext_vector_type(8)))  float  v8f;
typedef __attribute__((ext_vector_type(4)))  unsigned int v4u;
typedef __attribute__((ext_vector_type(8)))  int    v8i;
typedef __attribute__((ext_vector_type(4)))  int    v4i;

#if __has_builtin(__builtin_amdgcn_tensor_load_to_lds) && \
    __has_builtin(__builtin_amdgcn_s_wait_tensorcnt)
#define USE_TDM 1
#else
#define USE_TDM 0
#endif

#define DEV __device__ __forceinline__

DEV unsigned short f2bf(float f) {
  unsigned int u = __float_as_uint(f);
  unsigned int r = u + 0x7FFFu + ((u >> 16) & 1u);   // round-to-nearest-even
  return (unsigned short)(r >> 16);
}

// ---------------------------------------------------------------------------
// Problem constants
// ---------------------------------------------------------------------------
static constexpr int  BATCH   = 16;
static constexpr int  SEQ     = 1024;
static constexpr long MROWS   = (long)BATCH * SEQ;    // 16384
static constexpr int  DMODEL  = 512;
static constexpr int  DLOW    = 256;
static constexpr int  DMH     = 258;                  // 512/2 + 2
static constexpr int  DINNER  = 516;                  // 2*258
static constexpr int  DTRANK  = 17;
static constexpr int  DSTATE  = 16;
static constexpr int  NSPEC   = 528;                  // 514 padded to 16
static constexpr int  KH      = 288;                  // 258 padded to 32
static constexpr int  KDI     = 544;                  // 516 padded to 32
static constexpr int  NXD     = 64;                   // 49 padded
static constexpr int  KDT     = 32;                   // 17 padded
static constexpr int  NYH     = 272;                  // 258 padded to 16
static constexpr int  KCAT    = 544;                  // 514 padded to 32

static constexpr int FLAG_ACC  = 1;
static constexpr int FLAG_SILU = 2;

// ---------------------------------------------------------------------------
// WMMA GEMM:  C[M x N] (f32) = A[M x K] (bf16, row-major, ld=lda)
//                            x W[N x K]^T (bf16, row-major N-major, ld=ldw)
//            (+ bias[col]) (+= C if FLAG_ACC) (silu if FLAG_SILU)
// Tiles: block 128x128, K-step 32, 256 threads = 8 waves (4x2),
// each wave computes 32x64 = 2x4 grid of v_wmma_f32_16x16x32_bf16.
// Tile staging: Tensor Data Mover (tensor_load_to_lds, TENSORcnt), with
// pad_enable reproducing the 64B-row + 16B-pad LDS layout and tensor_dim1
// providing free zero-fill past the N edge.
// ---------------------------------------------------------------------------
#define BM 128
#define BN 128
#define BK 32
#define LDSP 40   // BK + 8 ushort pad -> 80B row stride, conflict-free b128 reads

union Frag { uint4 u[2]; v16bf v; };

#if USE_TDM
// One 2-D TDM descriptor: tile = rows x 16 dwords (= 128 x 64B), dword stride
// ld_elems/2, destination LDS rows padded 64B + 16B.
DEV void tdm_load_tile(const unsigned short* gbase, long row0, int ld_elems,
                       int kbase, int rows_valid, unsigned lds_byte_addr) {
  unsigned long long ga =
      (unsigned long long)(uintptr_t)(gbase + row0 * (long)ld_elems + kbase);
  v4u g0;
  g0[0] = 1u;                                   // count=1 (valid user D#)
  g0[1] = lds_byte_addr;                        // LDS byte address
  g0[2] = (unsigned)(ga & 0xFFFFFFFFu);         // global_addr[31:0]
  g0[3] = (unsigned)((ga >> 32) & 0x01FFFFFFu)  // global_addr[56:32]
        | (2u << 30);                           // type = 2 ("image")

  const unsigned td0 = 16u;                     // tensor_dim0: 16 dwords / row
  const unsigned td1 = (unsigned)rows_valid;    // tensor_dim1: OOB rows -> 0
  const unsigned st0 = (unsigned)(ld_elems >> 1); // dim0 stride in dwords
  v8i g1;
  g1[0] = (int)((2u << 16)                      // data_size = 4B
              | (1u << 20)                      // pad_enable
              | (3u << 22)                      // pad_interval: 16 dwords (64B)
              | (3u << 25));                    // pad_amount  : 4 dwords (16B)
  g1[1] = (int)((td0 & 0xFFFFu) << 16);         // tensor_dim0[15:0] @ bit48
  g1[2] = (int)((td0 >> 16) | ((td1 & 0xFFFFu) << 16));
  g1[3] = (int)((td1 >> 16) | (16u << 16));     // tile_dim0 = 16 dwords
  g1[4] = (int)(128u);                          // tile_dim1 = 128, tile_dim2 = 0
  g1[5] = (int)st0;                             // tensor_dim0_stride[31:0]
  g1[6] = 0;
  g1[7] = 0;
  v4i z4 = {0, 0, 0, 0};
  v8i z8 = {0, 0, 0, 0, 0, 0, 0, 0};
  __builtin_amdgcn_tensor_load_to_lds(g0, g1, z4, z4, z8, 0);
}
#endif

__global__ __launch_bounds__(256) void k_gemm(
    const unsigned short* __restrict__ A, int lda,
    const unsigned short* __restrict__ W, int ldw,
    float* __restrict__ C, int ldc,
    int N, int KT, const float* __restrict__ bias, int flags)
{
  __shared__ unsigned short As[2][BM][LDSP];
  __shared__ unsigned short Bs[2][BN][LDSP];

  const int tid  = threadIdx.x;
  const int lane = tid & 31;
  const int wave = tid >> 5;
  const int wm   = wave >> 1;          // 0..3 : wave row
  const int wn   = wave & 1;           // 0..1 : wave col
  const int half = lane >> 4;          // 0/1
  const int lrow = lane & 15;
  const long m0  = (long)blockIdx.x * BM;
  const int  n0  = blockIdx.y * BN;

  v8f acc[2][4];
#pragma unroll
  for (int i = 0; i < 2; ++i)
#pragma unroll
    for (int j = 0; j < 4; ++j)
#pragma unroll
      for (int e = 0; e < 8; ++e) acc[i][j][e] = 0.f;

#if USE_TDM
  auto issue = [&](int buf, int kt) {
    if (wave == 0)
      tdm_load_tile(A, m0, lda, kt * BK, BM,
                    (unsigned)(uintptr_t)&As[buf][0][0]);
    else if (wave == 1)
      tdm_load_tile(W, n0, ldw, kt * BK, N - n0,
                    (unsigned)(uintptr_t)&Bs[buf][0][0]);
  };
#else
  uint4 ra[2], rb[2];
  auto loadG = [&](int kbase) {
#pragma unroll
    for (int i = 0; i < 2; ++i) {
      int c = tid + i * 256;
      int r = c >> 2, kc = (c & 3) << 3;          // 8-ushort (16B) chunks
      ra[i] = *(const uint4*)(A + (m0 + r) * lda + kbase + kc);
      int n = n0 + r;
      if (n < N) rb[i] = *(const uint4*)(W + (long)n * ldw + kbase + kc);
      else       rb[i] = uint4{0u, 0u, 0u, 0u};
    }
  };
  auto storeL = [&](int buf) {
#pragma unroll
    for (int i = 0; i < 2; ++i) {
      int c = tid + i * 256;
      int r = c >> 2, kc = (c & 3) << 3;
      *(uint4*)&As[buf][r][kc] = ra[i];
      *(uint4*)&Bs[buf][r][kc] = rb[i];
    }
  };
#endif

  auto compute = [&](int buf) {
    Frag a[2], b[4];
#pragma unroll
    for (int mt = 0; mt < 2; ++mt) {
      const unsigned short* base = &As[buf][wm * 32 + mt * 16 + lrow][0];
      // A 16x32 bf16 lane layout: lane half h holds K in [8h,8h+8) U [16+8h,16+8h+8)
      a[mt].u[0] = *(const uint4*)(base + 8 * half);
      a[mt].u[1] = *(const uint4*)(base + 16 + 8 * half);
    }
#pragma unroll
    for (int nt = 0; nt < 4; ++nt) {
      const unsigned short* base = &Bs[buf][wn * 64 + nt * 16 + lrow][0];
      // B 32x16 bf16 lane layout: lane n, K in [8h,8h+8) U [16+8h,16+8h+8) (N-major LDS)
      b[nt].u[0] = *(const uint4*)(base + 8 * half);
      b[nt].u[1] = *(const uint4*)(base + 16 + 8 * half);
    }
#pragma unroll
    for (int mt = 0; mt < 2; ++mt)
#pragma unroll
      for (int nt = 0; nt < 4; ++nt)
        acc[mt][nt] = __builtin_amdgcn_wmma_f32_16x16x32_bf16(
            false, a[mt].v, false, b[nt].v, (short)0, acc[mt][nt], false, false);
  };

#if USE_TDM
  issue(0, 0);
  if (wave < 2) __builtin_amdgcn_s_wait_tensorcnt(0);
  __syncthreads();
  for (int kt = 0; kt < KT; ++kt) {
    const int cur = kt & 1;
    if (kt + 1 < KT) issue(cur ^ 1, kt + 1);   // DMA next tiles while computing
    compute(cur);
    if (kt + 1 < KT && wave < 2) __builtin_amdgcn_s_wait_tensorcnt(0);
    __syncthreads();
  }
#else
  loadG(0);
  storeL(0);
  __syncthreads();
  for (int kt = 0; kt < KT; ++kt) {
    const int cur = kt & 1;
    if (kt + 2 < KT) {  // L2 prefetch of the tile after next (global_prefetch_b8)
      __builtin_prefetch((const void*)(A + (m0 + (tid >> 2)) * lda + (kt + 2) * BK), 0, 1);
      if (n0 + (tid >> 2) < N)
        __builtin_prefetch((const void*)(W + (long)(n0 + (tid >> 2)) * ldw + (kt + 2) * BK), 0, 1);
    }
    if (kt + 1 < KT) loadG((kt + 1) * BK);
    compute(cur);
    if (kt + 1 < KT) storeL(cur ^ 1);
    __syncthreads();
  }
#endif

  // Epilogue. D layout: VGPR v -> row = base + v + 8*half, col = base + lrow.
  const bool facc  = (flags & FLAG_ACC) != 0;
  const bool fsilu = (flags & FLAG_SILU) != 0;
#pragma unroll
  for (int nt = 0; nt < 4; ++nt) {
    const int col = n0 + wn * 64 + nt * 16 + lrow;
    if (col >= N) continue;
    const float bv = bias ? bias[col] : 0.f;
#pragma unroll
    for (int mt = 0; mt < 2; ++mt) {
      const long rbase = m0 + wm * 32 + mt * 16 + 8 * half;
#pragma unroll
      for (int v = 0; v < 8; ++v) {
        float val = acc[mt][nt][v] + bv;
        const long idx = (rbase + v) * ldc + col;
        if (facc)  val += C[idx];
        if (fsilu) val = val / (1.f + __expf(-val));
        C[idx] = val;
      }
    }
  }
}

// ---------------------------------------------------------------------------
// Weight / basis preparation kernels
// ---------------------------------------------------------------------------

// Forward DFT basis, laid out N-major (row c = output column, K = 512 time):
// cols 0..127  = Re bins 0..127, 128..255 = Im bins 0..127   (x_l_cat)
// cols 256..384= Re bins 128..256, 385..513 = Im bins 128..256 (x_h_cat)
__global__ void k_gen_fdft(unsigned short* __restrict__ F) {
  long gid = (long)blockIdx.x * blockDim.x + threadIdx.x;
  if (gid >= (long)NSPEC * DMODEL) return;
  int c = (int)(gid / DMODEL), n = (int)(gid % DMODEL);
  float val = 0.f;
  if (c < 514) {
    int bin; bool re;
    if (c < 128)      { bin = c;        re = true;  }
    else if (c < 256) { bin = c - 128;  re = false; }
    else if (c < 385) { bin = c - 128;  re = true;  }
    else              { bin = c - 257;  re = false; }
    float ang = 6.2831853071795864769f * (float)(((long)n * bin) % DMODEL) / (float)DMODEL;
    float s = 0.044194173824159220275f;   // 1/sqrt(512)
    val = re ? s * __cosf(ang) : -s * __sinf(ang);
  }
  F[gid] = f2bf(val);
}

// Inverse DFT basis, N-major (row n = time sample, K = 544 interleaved Re/Im):
// flat component c: bin k = c>>1, real if !(c&1); scale s_k/sqrt(512), s_k = 1 for
// k in {0,256}, else 2 (Hermitian symmetry of irfft).
__global__ void k_gen_finv(unsigned short* __restrict__ F) {
  long gid = (long)blockIdx.x * blockDim.x + threadIdx.x;
  if (gid >= (long)DMODEL * KCAT) return;
  int n = (int)(gid / KCAT), c = (int)(gid % KCAT);
  float val = 0.f;
  if (c < 514) {
    int  k  = c >> 1;
    bool re = (c & 1) == 0;
    float sk = (k == 0 || k == 256) ? 1.f : 2.f;
    float w  = sk * 0.044194173824159220275f;
    float ang = 6.2831853071795864769f * (float)(((long)k * n) % DMODEL) / (float)DMODEL;
    val = re ? w * __cosf(ang) : -w * __sinf(ang);
  }
  F[gid] = f2bf(val);
}

// taylor_coeffs (u,i,o=3) -> W1[u][i]=c[u,i,1], W2[u][i]=c[u,i,2],
// bias0[u] = taylor_bias[u] + sum_i c[u,i,0]    (order-0 term: x^0 == 1)
__global__ void k_prep_taylor(const float* __restrict__ coeffs,
                              const float* __restrict__ tbias,
                              unsigned short* __restrict__ W1,
                              unsigned short* __restrict__ W2,
                              float* __restrict__ bias0) {
  int u = threadIdx.x;   // one block of 256
  float s = tbias[u];
  for (int i = 0; i < DLOW; ++i) {
    long base = ((long)u * DLOW + i) * 3;
    s += coeffs[base + 0];
    W1[(long)u * DLOW + i] = f2bf(coeffs[base + 1]);
    W2[(long)u * DLOW + i] = f2bf(coeffs[base + 2]);
  }
  bias0[u] = s;
}

// Generic weight cast + K-pad:  dst[n][0..Kp) = bf16(src[n][0..K)), pad 0
__global__ void k_prep_w(unsigned short* __restrict__ dst,
                         const float* __restrict__ src,
                         int N, int K, int Kp) {
  long gid = (long)blockIdx.x * blockDim.x + threadIdx.x;
  if (gid >= (long)N * Kp) return;
  int n = (int)(gid / Kp), k = (int)(gid % Kp);
  dst[gid] = (k < K) ? f2bf(src[(long)n * K + k]) : (unsigned short)0;
}

// ---------------------------------------------------------------------------
// Elementwise / pipeline kernels
// ---------------------------------------------------------------------------
__global__ void k_cvt_x(const float* __restrict__ x, unsigned short* __restrict__ o) {
  long gid = (long)blockIdx.x * blockDim.x + threadIdx.x;
  if (gid < MROWS * DMODEL) o[gid] = f2bf(x[gid]);
}

// spec (M x 528 f32) -> xl (M x 256 bf16), xl2 = xl^2, xh (M x 288 bf16 zero-pad)
__global__ void k_split_spec(const float* __restrict__ spec,
                             unsigned short* __restrict__ xl,
                             unsigned short* __restrict__ xl2,
                             unsigned short* __restrict__ xh) {
  long gid = (long)blockIdx.x * blockDim.x + threadIdx.x;
  if (gid >= MROWS * KH) return;
  long r = gid / KH; int c = (int)(gid % KH);
  xh[r * KH + c] = (c < DMH) ? f2bf(spec[r * NSPEC + 256 + c]) : (unsigned short)0;
  if (c < DLOW) {
    float v = spec[r * NSPEC + c];
    xl [r * DLOW + c] = f2bf(v);
    xl2[r * DLOW + c] = f2bf(v * v);
  }
}

// depthwise causal conv (D_CONV=4) + SiLU -> bf16 (K-padded 544) and f32 copies
__global__ void k_conv_silu(const float* __restrict__ xm, // M x 516
                            const float* __restrict__ cw, // 516 x 4
                            const float* __restrict__ cb, // 516
                            unsigned short* __restrict__ xmbf, // M x 544
                            float* __restrict__ xms) {          // M x 516
  long gid = (long)blockIdx.x * blockDim.x + threadIdx.x;
  if (gid >= MROWS * KDI) return;
  long r = gid / KDI; int d = (int)(gid % KDI);
  if (d >= DINNER) { xmbf[r * KDI + d] = 0; return; }
  int b = (int)(r >> 10), l = (int)(r & 1023);
  float a = cb[d];
#pragma unroll
  for (int j = 0; j < 4; ++j) {
    int ls = l - 3 + j;
    if (ls >= 0) a += xm[((long)b * SEQ + ls) * DINNER + d] * cw[d * 4 + j];
  }
  float s = a / (1.f + __expf(-a));
  xmbf[r * KDI + d] = f2bf(s);
  xms [r * DINNER + d] = s;
}

// dt columns of x_dbl -> bf16 K-padded to 32
__global__ void k_dta(const float* __restrict__ xdbl, unsigned short* __restrict__ dtA) {
  long gid = (long)blockIdx.x * blockDim.x + threadIdx.x;
  if (gid >= MROWS * KDT) return;
  long r = gid >> 5; int c = (int)(gid & 31);
  dtA[gid] = (c < DTRANK) ? f2bf(xdbl[r * NXD + c]) : (unsigned short)0;
}

__global__ void k_softplus(float* __restrict__ dt, const float* __restrict__ b) {
  long gid = (long)blockIdx.x * blockDim.x + threadIdx.x;
  if (gid >= MROWS * DINNER) return;
  int d = (int)(gid % DINNER);
  float v = dt[gid] + b[d];
  dt[gid] = (v > 20.f) ? v : log1pf(__expf(v));
}

// Selective scan: one thread per (batch, channel); 16 states in registers.
// yA = ((scan_y + x*D) * silu_z) as bf16 into a K-padded (544) buffer.
__global__ void k_scan(const float* __restrict__ dt,    // M x 516
                       const float* __restrict__ xms,   // M x 516
                       const float* __restrict__ xdbl,  // M x 64  (B:17..32, C:33..48)
                       const float* __restrict__ zs,    // M x 516 (silu'ed z)
                       const float* __restrict__ Alog,  // 516 x 16
                       const float* __restrict__ Dp,    // 516
                       unsigned short* __restrict__ yA) // M x 544
{
  int idx = blockIdx.x * blockDim.x + threadIdx.x;
  if (idx >= BATCH * KDI) return;
  int b = idx / KDI, d = idx % KDI;
  if (d >= DINNER) {  // deterministically zero the K-pad columns
    for (int l = 0; l < SEQ; ++l) yA[((long)b * SEQ + l) * KDI + d] = 0;
    return;
  }
  float Ad[DSTATE];
#pragma unroll
  for (int s = 0; s < DSTATE; ++s) Ad[s] = -__expf(Alog[d * DSTATE + s]);
  const float Dv = Dp[d];
  float h[DSTATE];
#pragma unroll
  for (int s = 0; s < DSTATE; ++s) h[s] = 0.f;

  for (int l = 0; l < SEQ; ++l) {
    long r = (long)b * SEQ + l;
    float dtv = dt [r * DINNER + d];
    float xt  = xms[r * DINNER + d];
    const float* bc = xdbl + r * NXD;
    float dx = dtv * xt;
    float y = 0.f;
#pragma unroll
    for (int s = 0; s < DSTATE; ++s) {
      float dA = __expf(dtv * Ad[s]);
      h[s] = dA * h[s] + dx * bc[DTRANK + s];
      y += h[s] * bc[DTRANK + DSTATE + s];
    }
    float out = (y + xt * Dv) * zs[r * DINNER + d];
    yA[r * KDI + d] = f2bf(out);
  }
}

// concat y_l (256) + y_h (258) -> interleaved-complex cat, bf16, K-padded 544
__global__ void k_build_ycat(const float* __restrict__ yl,  // M x 256
                             const float* __restrict__ yh,  // M x 272 (258 valid)
                             unsigned short* __restrict__ ycat) {
  long gid = (long)blockIdx.x * blockDim.x + threadIdx.x;
  if (gid >= MROWS * KCAT) return;
  long r = gid / KCAT; int c = (int)(gid % KCAT);
  float v = 0.f;
  if (c < DLOW)       v = yl[r * DLOW + c];
  else if (c < 514)   v = yh[r * NYH + (c - DLOW)];
  ycat[gid] = f2bf(v);
}

// RMS-norm over 512 + residual. One block (256 threads) per row.
__global__ __launch_bounds__(256) void k_rmsnorm(const float* __restrict__ y,
                                                 const float* __restrict__ x,
                                                 const float* __restrict__ nw,
                                                 float* __restrict__ out) {
  __shared__ float red[256];
  long r = blockIdx.x;
  int  t = threadIdx.x;
  float v0 = y[r * DMODEL + t];
  float v1 = y[r * DMODEL + 256 + t];
  red[t] = v0 * v0 + v1 * v1;
  __syncthreads();
  for (int s = 128; s > 0; s >>= 1) {
    if (t < s) red[t] += red[t + s];
    __syncthreads();
  }
  float scale = rsqrtf(red[0] * (1.f / DMODEL) + 1e-5f);
  out[r * DMODEL + t]       = v0 * scale * nw[t]       + x[r * DMODEL + t];
  out[r * DMODEL + 256 + t] = v1 * scale * nw[256 + t] + x[r * DMODEL + 256 + t];
}

// ---------------------------------------------------------------------------
// Host launch
// ---------------------------------------------------------------------------
static inline int cdiv(long a, int b) { return (int)((a + b - 1) / b); }

extern "C" void kernel_launch(void* const* d_in, const int* in_sizes, int n_in,
                              void* d_out, int out_size, void* d_ws, size_t ws_size,
                              hipStream_t stream) {
  (void)in_sizes; (void)n_in; (void)out_size; (void)ws_size;
  const float* x        = (const float*)d_in[0];
  const float* tcoef    = (const float*)d_in[1];
  const float* tbias    = (const float*)d_in[2];
  const float* in_proj  = (const float*)d_in[3];
  const float* conv_w   = (const float*)d_in[4];
  const float* conv_b   = (const float*)d_in[5];
  const float* x_proj   = (const float*)d_in[6];
  const float* dt_proj  = (const float*)d_in[7];
  const float* dt_bias  = (const float*)d_in[8];
  const float* A_log    = (const float*)d_in[9];
  const float* D_param  = (const float*)d_in[10];
  const float* out_proj = (const float*)d_in[11];
  const float* norm_w   = (const float*)d_in[12];
  float* out = (float*)d_out;

  // -- workspace arena with time-based aliasing ------------------------------
  char* base = (char*)d_ws;
  size_t off = 0;
  auto alloc = [&](size_t bytes) -> char* {
    char* r = base + off;
    off = (off + bytes + 255) & ~(size_t)255;
    return r;
  };
  // persistent weight/basis buffers
  unsigned short* Fdft  = (unsigned short*)alloc((size_t)NSPEC * DMODEL * 2);
  unsigned short* Finv  = (unsigned short*)alloc((size_t)DMODEL * KCAT * 2);
  unsigned short* W1    = (unsigned short*)alloc((size_t)DLOW * DLOW * 2);
  unsigned short* W2    = (unsigned short*)alloc((size_t)DLOW * DLOW * 2);
  float*          bias0 = (float*)alloc(DLOW * 4);
  unsigned short* Winp  = (unsigned short*)alloc((size_t)(2 * DINNER) * KH * 2);
  unsigned short* Wxp   = (unsigned short*)alloc((size_t)49 * KDI * 2);
  unsigned short* Wdtp  = (unsigned short*)alloc((size_t)DINNER * KDT * 2);
  unsigned short* Woutp = (unsigned short*)alloc((size_t)DMH * KDI * 2);
  // big regions (aliased across pipeline phases; liveness is disjoint)
  char* R1 = alloc((size_t)MROWS * DMODEL * 4);         // Abf(bf16) -> y_l(f32)
  char* R2 = alloc((size_t)MROWS * NSPEC * 4);          // spec(f32) -> xms(f32)
  char* R3 = alloc((size_t)MROWS * (DLOW*2 + DLOW*2 + KH*2)); // xl|xl2|xh -> y_h
  char* R4 = alloc((size_t)MROWS * DINNER * 4);         // xm(f32) -> dt(f32)
  char* R5 = alloc((size_t)MROWS * DINNER * 4);         // zs(f32) -> yout(f32)
  char* R6 = alloc((size_t)MROWS * KDI * 2);            // xm_bf -> yA
  char* R7 = alloc((size_t)MROWS * KCAT * 2);           // ycat
  char* R8 = alloc((size_t)MROWS * NXD * 4);            // x_dbl
  char* R9 = alloc((size_t)MROWS * KDT * 2);            // dtA

  unsigned short* Abf  = (unsigned short*)R1;
  float*          y_l  = (float*)R1;
  float*          spec = (float*)R2;
  float*          xms  = (float*)R2;
  unsigned short* xl   = (unsigned short*)R3;
  unsigned short* xl2  = (unsigned short*)(R3 + (size_t)MROWS * DLOW * 2);
  unsigned short* xh   = (unsigned short*)(R3 + (size_t)MROWS * DLOW * 4);
  float*          y_h  = (float*)R3;
  float*          xm   = (float*)R4;
  float*          dt   = (float*)R4;
  float*          zs   = (float*)R5;
  float*          yout = (float*)R5;
  unsigned short* xmbf = (unsigned short*)R6;
  unsigned short* yA   = (unsigned short*)R6;
  unsigned short* ycat = (unsigned short*)R7;
  float*          xdbl = (float*)R8;
  unsigned short* dtA  = (unsigned short*)R9;

  const int T = 256;

  // ---- prep ----------------------------------------------------------------
  k_gen_fdft  <<<cdiv((long)NSPEC * DMODEL, T), T, 0, stream>>>(Fdft);
  k_gen_finv  <<<cdiv((long)DMODEL * KCAT, T), T, 0, stream>>>(Finv);
  k_prep_taylor<<<1, 256, 0, stream>>>(tcoef, tbias, W1, W2, bias0);
  k_prep_w<<<cdiv((long)(2 * DINNER) * KH, T), T, 0, stream>>>(Winp, in_proj, 2 * DINNER, DMH, KH);
  k_prep_w<<<cdiv((long)49 * KDI, T), T, 0, stream>>>(Wxp, x_proj, 49, DINNER, KDI);
  k_prep_w<<<cdiv((long)DINNER * KDT, T), T, 0, stream>>>(Wdtp, dt_proj, DINNER, DTRANK, KDT);
  k_prep_w<<<cdiv((long)DMH * KDI, T), T, 0, stream>>>(Woutp, out_proj, DMH, DINNER, KDI);
  k_cvt_x<<<cdiv(MROWS * DMODEL, T), T, 0, stream>>>(x, Abf);

  // ---- forward DFT as bf16 WMMA GEMM: spec = x @ Fdft^T --------------------
  k_gemm<<<dim3(MROWS / BM, cdiv(NSPEC, BN)), T, 0, stream>>>(
      Abf, DMODEL, Fdft, DMODEL, spec, NSPEC, NSPEC, DMODEL / BK, nullptr, 0);
  k_split_spec<<<cdiv(MROWS * KH, T), T, 0, stream>>>(spec, xl, xl2, xh);

  // ---- Taylor-KAN: y_l = x@W1^T + bias0, then += x^2 @ W2^T ----------------
  k_gemm<<<dim3(MROWS / BM, cdiv(DLOW, BN)), T, 0, stream>>>(
      xl, DLOW, W1, DLOW, y_l, DLOW, DLOW, DLOW / BK, bias0, 0);
  k_gemm<<<dim3(MROWS / BM, cdiv(DLOW, BN)), T, 0, stream>>>(
      xl2, DLOW, W2, DLOW, y_l, DLOW, DLOW, DLOW / BK, nullptr, FLAG_ACC);

  // ---- Mamba: in_proj (xm part plain, z part fused-SiLU) -------------------
  k_gemm<<<dim3(MROWS / BM, cdiv(DINNER, BN)), T, 0, stream>>>(
      xh, KH, Winp, KH, xm, DINNER, DINNER, KH / BK, nullptr, 0);
  k_gemm<<<dim3(MROWS / BM, cdiv(DINNER, BN)), T, 0, stream>>>(
      xh, KH, Winp + (size_t)DINNER * KH, KH, zs, DINNER, DINNER, KH / BK, nullptr, FLAG_SILU);

  k_conv_silu<<<cdiv(MROWS * KDI, T), T, 0, stream>>>(xm, conv_w, conv_b, xmbf, xms);

  k_gemm<<<dim3(MROWS / BM, 1), T, 0, stream>>>(
      xmbf, KDI, Wxp, KDI, xdbl, NXD, 49, KDI / BK, nullptr, 0);

  k_dta<<<cdiv(MROWS * KDT, T), T, 0, stream>>>(xdbl, dtA);
  k_gemm<<<dim3(MROWS / BM, cdiv(DINNER, BN)), T, 0, stream>>>(
      dtA, KDT, Wdtp, KDT, dt, DINNER, DINNER, KDT / BK, nullptr, 0);
  k_softplus<<<cdiv(MROWS * DINNER, T), T, 0, stream>>>(dt, dt_bias);

  k_scan<<<cdiv((long)BATCH * KDI, T), T, 0, stream>>>(dt, xms, xdbl, zs, A_log, D_param, yA);

  k_gemm<<<dim3(MROWS / BM, cdiv(DMH, BN)), T, 0, stream>>>(
      yA, KDI, Woutp, KDI, y_h, NYH, DMH, KDI / BK, nullptr, 0);

  // ---- inverse DFT GEMM + RMS-norm + residual ------------------------------
  k_build_ycat<<<cdiv(MROWS * KCAT, T), T, 0, stream>>>(y_l, y_h, ycat);
  k_gemm<<<dim3(MROWS / BM, cdiv(DMODEL, BN)), T, 0, stream>>>(
      ycat, KCAT, Finv, KCAT, yout, DMODEL, DMODEL, KCAT / BK, nullptr, 0);
  k_rmsnorm<<<(int)MROWS, 256, 0, stream>>>(yout, x, norm_w, out);
}
